// MistralAttentionForSVDKV_76519137346144
// MI455X (gfx1250) — matrix operand order repaired and checked by reading, not compiled, not run on hardware
//
#include <hip/hip_runtime.h>

// ---------------------------------------------------------------------------
// MistralAttentionForSVDKV — MI455X (gfx1250) implementation
// B=1, S=2048, H=4096, NH=32, KVH=8, HD=128, KC=VC=64, WIN=32, 4-bit fakequant
// BF16 WMMA (v_wmma_f32_16x16x32_bf16) for all GEMMs + flash attention.
// ---------------------------------------------------------------------------

typedef __bf16 bf16_t;
typedef __bf16 bf16x8 __attribute__((ext_vector_type(8)));
typedef __bf16 v16bf  __attribute__((ext_vector_type(16)));
typedef float  v8f    __attribute__((ext_vector_type(8)));

#define DEV __device__ __forceinline__

namespace cfg {
constexpr int S    = 2048;
constexpr int Hd   = 4096;   // hidden
constexpr int NH   = 32;
constexpr int KVH  = 8;
constexpr int HD   = 128;
constexpr int KC   = 64;
constexpr int VC   = 64;
constexpr int WIN  = 32;
constexpr int NQKV = Hd + KVH * HD + KVH * KC + KVH * HD + KVH * VC; // 7168
constexpr int KOFF = Hd;                    // 4096 : k_orig columns
constexpr int KCOF = Hd + KVH * HD;         // 5120 : k_comp columns
constexpr int VOFF = Hd + KVH * (HD + KC);  // 5632 : v_orig columns
constexpr int VCOF = VOFF + KVH * HD;       // 6656 : v_comp columns
}

// ----------------------------- helpers -------------------------------------

DEV unsigned short f2bf_bits(float f) {
  unsigned u = __builtin_bit_cast(unsigned, f);
  u += 0x7FFFu + ((u >> 16) & 1u);          // round-to-nearest-even
  return (unsigned short)(u >> 16);
}
DEV bf16_t f2bf(float f) { return __builtin_bit_cast(bf16_t, f2bf_bits(f)); }

// A-matrix fragment (16x32, row per lane pair): elements 0..7 at k0+8*half,
// elements 8..15 at k0+16+8*half (per CDNA5 ISA 16-bit A layout).
DEV v16bf load_afrag(const unsigned short* p) {
  bf16x8 lo = *reinterpret_cast<const bf16x8*>(p);
  bf16x8 hi = *reinterpret_cast<const bf16x8*>(p + 16);
  v16bf r;
#pragma unroll
  for (int i = 0; i < 8; ++i) { r[i] = lo[i]; r[i + 8] = hi[i]; }
  return r;
}

// B-matrix fragment (32x16, column per lane): elements 0..7 at k0+8*half,
// elements 8..15 at k0+16+8*half (same K order as the A fragment).
DEV v16bf load_bfrag(const unsigned short* p) {
  bf16x8 lo = *reinterpret_cast<const bf16x8*>(p);
  bf16x8 hi = *reinterpret_cast<const bf16x8*>(p + 16);
  v16bf r;
#pragma unroll
  for (int i = 0; i < 8; ++i) { r[i] = lo[i]; r[i + 8] = hi[i]; }
  return r;
}

DEV v8f wmma_bf16(v16bf a, v16bf b, v8f c) {
  return __builtin_amdgcn_wmma_f32_16x16x32_bf16(false, a, false, b,
                                                 (short)0, c, false, false);
}

// ----------------------------- conversions ---------------------------------

__global__ void k_f32_to_bf16(const float* __restrict__ in,
                              unsigned short* __restrict__ out, int n) {
  int i = blockIdx.x * blockDim.x + threadIdx.x;
  if (i < n) out[i] = f2bf_bits(in[i]);
}

// W [K x N] f32 row-major  ->  Wt [rowOff+N rows x ldk] bf16, Wt[n][k] = W[k][n]
__global__ void k_transpose_bf16(const float* __restrict__ W,
                                 unsigned short* __restrict__ Wt,
                                 int N, int rowOff, int ldk) {
  __shared__ float tile[32][33];
  int k0 = blockIdx.y * 32, n0 = blockIdx.x * 32;
  int tx = threadIdx.x, ty = threadIdx.y;      // 32 x 8
#pragma unroll
  for (int i = 0; i < 4; ++i)
    tile[ty + i * 8][tx] = W[(size_t)(k0 + ty + i * 8) * N + n0 + tx];
  __syncthreads();
#pragma unroll
  for (int i = 0; i < 4; ++i)
    Wt[(size_t)(rowOff + n0 + ty + i * 8) * ldk + k0 + tx] =
        f2bf_bits(tile[tx][ty + i * 8]);
}

// ----------------------------- GEMM (bf16 WMMA) ----------------------------
// C[M x N] f32 = A[M x K] bf16 row-major  x  Bt[N x K] bf16 (B column-major).
// Each wave: 32x64 tile (2 A-frags x 4 B-frags -> 8 wmma / K-step of 32).
// B fragments are reused across the two row tiles: 12 b128 loads per 8 wmma.
__global__ void k_gemm_bf16(const unsigned short* __restrict__ A,
                            const unsigned short* __restrict__ Bt,
                            float* __restrict__ C, int M, int N, int K) {
  const int lane = threadIdx.x & 31;
  const int widx = threadIdx.x >> 5;
  const int half = lane >> 4;
  const int ml   = lane & 15;
  const int nt64 = N >> 6;
  const int w    = blockIdx.x * (blockDim.x >> 5) + widx;
  const int mt   = w / nt64;
  const int nt   = w % nt64;
  if (mt * 32 >= M) return;
  const size_t m0 = (size_t)mt * 32, n0 = (size_t)nt * 64;

  v8f acc[2][4];
#pragma unroll
  for (int t = 0; t < 2; ++t)
#pragma unroll
    for (int j = 0; j < 4; ++j) acc[t][j] = (v8f)0.0f;

  const unsigned short* arow0 = A + (m0 + ml) * K;
  const unsigned short* arow1 = A + (m0 + 16 + ml) * K;
  const unsigned short* brow  = Bt + (n0 + ml) * K;

  for (int k0 = 0; k0 < K; k0 += 32) {
    __builtin_prefetch(arow0 + k0 + 256, 0, 3);  // global_prefetch_b8 (near)
    __builtin_prefetch(arow1 + k0 + 256, 0, 3);
    __builtin_prefetch(brow + k0 + 256, 0, 3);
    v16bf a0 = load_afrag(arow0 + k0 + 8 * half);
    v16bf a1 = load_afrag(arow1 + k0 + 8 * half);
#pragma unroll
    for (int j = 0; j < 4; ++j) {
      v16bf bf = load_bfrag(brow + (size_t)j * 16 * K + k0 + 8 * half);
      acc[0][j] = wmma_bf16(a0, bf, acc[0][j]);
      acc[1][j] = wmma_bf16(a1, bf, acc[1][j]);
    }
  }
  // D layout: element jj -> row jj+8*half, col = lane&15
#pragma unroll
  for (int t = 0; t < 2; ++t)
#pragma unroll
    for (int j = 0; j < 4; ++j)
#pragma unroll
      for (int jj = 0; jj < 8; ++jj)
        C[(m0 + t * 16 + jj + 8 * half) * N + n0 + j * 16 + ml] = acc[t][j][jj];
}

// ----------------------------- RoPE + split --------------------------------
// qkv f32 [S x 7168] -> q_bf [NH][S][HD], k_bf [KVH][S][HD], vT_bf [KVH][HD][S]
__global__ void k_rope_split(const float* __restrict__ qkv,
                             const int* __restrict__ pids,
                             unsigned short* __restrict__ qbf,
                             unsigned short* __restrict__ kbf,
                             unsigned short* __restrict__ vTbf) {
  using namespace cfg;
  const int s = blockIdx.x;
  const int d = threadIdx.x;                 // 0..127
  const int i = d & 63;
  const float pos = (float)pids[s];
  const float inv = __powf(10000.0f, -(float)(2 * i) / (float)HD);
  const float fr  = pos * inv;
  const float c = __cosf(fr), sn = __sinf(fr);
  const float* row = qkv + (size_t)s * NQKV;

#pragma unroll 4
  for (int h = 0; h < NH; ++h) {
    const float* b = row + h * HD;
    float x   = b[d];
    float rot = (d < 64) ? -b[d + 64] : b[d - 64];
    qbf[((size_t)h * S + s) * HD + d] = f2bf_bits(x * c + rot * sn);
  }
#pragma unroll 4
  for (int h = 0; h < KVH; ++h) {
    const float* b = row + KOFF + h * HD;
    float x   = b[d];
    float rot = (d < 64) ? -b[d + 64] : b[d - 64];
    kbf[((size_t)h * S + s) * HD + d] = f2bf_bits(x * c + rot * sn);
    float v = row[VOFF + h * HD + d];
    vTbf[((size_t)h * HD + d) * S + s] = f2bf_bits(v);  // transposed store
  }
}

// ----------------------------- fake-quant kernels --------------------------
// k_comp: per-channel stats over 32-token windows (quant_channel)
__global__ void k_quant_k_channel(const float* __restrict__ qkv,
                                  float* __restrict__ outk) {
  using namespace cfg;
  const int g  = blockIdx.x;          // KVH * (S/WIN) = 512
  const int kh = g >> 6;
  const int w  = g & 63;
  const int d  = threadIdx.x;         // 0..63
  const float* base = qkv + (size_t)(w * WIN) * NQKV + KCOF + kh * KC + d;
  float mn = 3.0e38f, mx = -3.0e38f;
#pragma unroll 8
  for (int t = 0; t < WIN; ++t) {
    float x = base[(size_t)t * NQKV];
    mn = fminf(mn, x); mx = fmaxf(mx, x);
  }
  const float scale = fmaxf(mx - mn, 1e-5f) * (1.0f / 15.0f);
#pragma unroll 8
  for (int t = 0; t < WIN; ++t) {
    float x = base[(size_t)t * NQKV];
    float q = fminf(fmaxf(rintf((x - mn) / scale), 0.0f), 15.0f);
    outk[((size_t)kh * S + w * WIN + t) * KC + d] = q * scale + mn;
  }
}

// v_comp: per-token stats over the 64 channels (quant_token, group==VC)
__global__ void k_quant_v_token(const float* __restrict__ qkv,
                                float* __restrict__ outv) {
  using namespace cfg;
  const int g  = blockIdx.x;          // KVH * S = 16384
  const int kh = g >> 11;
  const int s  = g & 2047;
  const int d  = threadIdx.x;         // 0..63
  __shared__ float smn[64], smx[64];
  const float x = qkv[(size_t)s * NQKV + VCOF + kh * VC + d];
  smn[d] = x; smx[d] = x;
  __syncthreads();
#pragma unroll
  for (int o = 32; o > 0; o >>= 1) {
    if (d < o) { smn[d] = fminf(smn[d], smn[d + o]); smx[d] = fmaxf(smx[d], smx[d + o]); }
    __syncthreads();
  }
  const float mn = smn[0], mx = smx[0];
  const float scale = fmaxf(mx - mn, 1e-5f) * (1.0f / 15.0f);
  float q = fminf(fmaxf(rintf((x - mn) / scale), 0.0f), 15.0f);
  outv[((size_t)kh * S + s) * VC + d] = q * scale + mn;
}

// ----------------------------- flash attention -----------------------------
// One wave per (head, 16-row q tile). Causal, online softmax, bf16 WMMA.
__global__ void __launch_bounds__(256, 2)
k_flash_attn(const unsigned short* __restrict__ qbf,
             const unsigned short* __restrict__ kbf,
             const unsigned short* __restrict__ vTbf,
             unsigned short* __restrict__ obf) {
  using namespace cfg;
  constexpr int WAVES = 8;
  __shared__ float s_lds[WAVES][16][33];   // score transpose per wave
  __shared__ float bcast[WAVES][16];       // per-row alpha / 1/l broadcast

  const int lane = threadIdx.x & 31;
  const int widx = threadIdx.x >> 5;
  const int half = lane >> 4;
  const int ml   = lane & 15;
  const int w    = blockIdx.x * WAVES + widx;
  const int head = w >> 7;                 // 128 q-tiles per head
  const int qt   = w & 127;
  if (head >= NH) return;
  const int q0   = qt * 16;
  const int kvh  = head >> 2;              // GQA group of 4

  // Q fragments: 4 x (16x32 over d)
  v16bf qfrag[4];
  {
    const unsigned short* qrow =
        qbf + ((size_t)head * S + q0 + ml) * HD;
#pragma unroll
    for (int dd = 0; dd < 4; ++dd)
      qfrag[dd] = load_afrag(qrow + dd * 32 + 8 * half);
  }

  v8f oacc[8];
#pragma unroll
  for (int n = 0; n < 8; ++n) oacc[n] = (v8f)0.0f;
  float mrun = -3.0e38f, lrun = 0.0f;      // stats for row (q0+ml), dup halves

  const unsigned short* kbase = kbf + (size_t)kvh * S * HD;
  const unsigned short* vbase = vTbf + (size_t)kvh * HD * S;
  const float scale = 0.088388347648318447f;   // 1/sqrt(128)
  const int rowg = q0 + ml;
  const int nblk = ((q0 + 15) >> 5) + 1;       // causal 32-key blocks

  for (int kb = 0; kb < nblk; ++kb) {
    const int key0 = kb * 32;
    // ---- scores: two 16x16 D tiles over 32 keys ----
#pragma unroll
    for (int t = 0; t < 2; ++t) {
      v8f sacc = (v8f)0.0f;
      const unsigned short* krow = kbase + (size_t)(key0 + t * 16 + ml) * HD;
#pragma unroll
      for (int dd = 0; dd < 4; ++dd) {
        v16bf kf = load_bfrag(krow + dd * 32 + 8 * half);
        sacc = wmma_bf16(qfrag[dd], kf, sacc);
      }
#pragma unroll
      for (int j = 0; j < 8; ++j)
        s_lds[widx][j + 8 * half][t * 16 + ml] = sacc[j];
    }
    // ---- transpose to row-per-lane, mask, online softmax ----
    float sv[16];
    float mx = -3.0e38f;
#pragma unroll
    for (int e = 0; e < 16; ++e) {
      const int kk = (e < 8) ? (e + 8 * half) : (e + 8 + 8 * half);
      float x = s_lds[widx][ml][kk] * scale;
      x = ((key0 + kk) <= rowg) ? x : -3.0e38f;
      sv[e] = x;
      mx = fmaxf(mx, x);
    }
    mx = fmaxf(mx, __shfl_xor(mx, 16));
    const float mnew  = fmaxf(mrun, mx);
    const float alpha = __expf(mrun - mnew);
    float psum = 0.0f;
    v16bf pfrag;                              // already in A-frag layout
#pragma unroll
    for (int e = 0; e < 16; ++e) {
      float p = __expf(sv[e] - mnew);
      psum += p;
      pfrag[e] = f2bf(p);
    }
    psum += __shfl_xor(psum, 16);
    lrun = lrun * alpha + psum;
    mrun = mnew;
    // broadcast alpha into D-layout rows via LDS (wave-local, LDS in-order)
    if (half == 0) bcast[widx][ml] = alpha;
    float al[8];
#pragma unroll
    for (int j = 0; j < 8; ++j) al[j] = bcast[widx][j + 8 * half];
#pragma unroll
    for (int n = 0; n < 8; ++n)
#pragma unroll
      for (int j = 0; j < 8; ++j) oacc[n][j] *= al[j];
    // ---- P x V : 8 wmma over the 128-dim ----
#pragma unroll
    for (int n = 0; n < 8; ++n) {
      v16bf vf = load_bfrag(vbase + (size_t)(n * 16 + ml) * S + key0 + 8 * half);
      oacc[n] = wmma_bf16(pfrag, vf, oacc[n]);
    }
  }

  // ---- normalize by 1/l and store bf16 [S x NH*HD] ----
  if (half == 0) bcast[widx][ml] = 1.0f / lrun;
  float rl[8];
#pragma unroll
  for (int j = 0; j < 8; ++j) rl[j] = bcast[widx][j + 8 * half];
#pragma unroll
  for (int n = 0; n < 8; ++n)
#pragma unroll
    for (int j = 0; j < 8; ++j)
      obf[(size_t)(q0 + j + 8 * half) * (NH * HD) + head * HD + n * 16 + ml] =
          f2bf_bits(oacc[n][j] * rl[j]);
}

// ----------------------------- launch --------------------------------------

extern "C" void kernel_launch(void* const* d_in, const int* in_sizes, int n_in,
                              void* d_out, int out_size, void* d_ws,
                              size_t ws_size, hipStream_t stream) {
  using namespace cfg;
  (void)in_sizes; (void)n_in; (void)out_size; (void)ws_size;

  const float* hs   = (const float*)d_in[0];
  const int*   pids = (const int*)d_in[1];
  const float* Wq   = (const float*)d_in[2];
  const float* Wk_a = (const float*)d_in[3];
  const float* Wv_a = (const float*)d_in[4];
  const float* Wo   = (const float*)d_in[5];

  float* out_attn = (float*)d_out;                          // [S, H]
  float* out_kq   = out_attn + (size_t)S * Hd;              // [KVH, S, KC]
  float* out_vq   = out_kq + (size_t)KVH * S * KC;          // [KVH, S, VC]

  char* ws = (char*)d_ws;
  size_t off = 0;
  auto alloc = [&](size_t bytes) {
    char* p = ws + off;
    off += (bytes + 255) & ~(size_t)255;
    return p;
  };
  unsigned short* hs_bf  = (unsigned short*)alloc((size_t)S * Hd * 2);
  unsigned short* Wcat_t = (unsigned short*)alloc((size_t)NQKV * Hd * 2);
  unsigned short* Wo_t   = (unsigned short*)alloc((size_t)Hd * Hd * 2);
  float*          qkv    = (float*)alloc((size_t)S * NQKV * 4);
  unsigned short* q_bf   = (unsigned short*)alloc((size_t)NH * S * HD * 2);
  unsigned short* k_bf   = (unsigned short*)alloc((size_t)KVH * S * HD * 2);
  unsigned short* vT_bf  = (unsigned short*)alloc((size_t)KVH * HD * S * 2);
  unsigned short* at_bf  = (unsigned short*)alloc((size_t)S * NH * HD * 2);

  // 1) hidden_states f32 -> bf16
  {
    int n = S * Hd;
    k_f32_to_bf16<<<(n + 255) / 256, 256, 0, stream>>>(hs, hs_bf, n);
  }
  // 2) weights: transpose + convert (column-major bf16 for B fragments)
  {
    dim3 blk(32, 8);
    k_transpose_bf16<<<dim3(Hd / 32, Hd / 32), blk, 0, stream>>>(
        Wq, Wcat_t, Hd, 0, Hd);
    k_transpose_bf16<<<dim3((KVH * (HD + KC)) / 32, Hd / 32), blk, 0, stream>>>(
        Wk_a, Wcat_t, KVH * (HD + KC), Hd, Hd);
    k_transpose_bf16<<<dim3((KVH * (HD + VC)) / 32, Hd / 32), blk, 0, stream>>>(
        Wv_a, Wcat_t, KVH * (HD + VC), Hd + KVH * (HD + KC), Hd);
    k_transpose_bf16<<<dim3(Hd / 32, Hd / 32), blk, 0, stream>>>(
        Wo, Wo_t, Hd, 0, Hd);
  }
  // 3) fused QKV GEMM: [S x Hd] x [Hd x NQKV] -> qkv f32  (32x64 wave tiles)
  {
    int tiles = (S / 32) * (NQKV / 64);                     // 7168
    k_gemm_bf16<<<tiles / 8, 256, 0, stream>>>(hs_bf, Wcat_t, qkv, S, NQKV, Hd);
  }
  // 4) RoPE + split into head-major bf16 buffers (V transposed)
  k_rope_split<<<S, HD, 0, stream>>>(qkv, pids, q_bf, k_bf, vT_bf);
  // 5) fake-quant outputs
  k_quant_k_channel<<<KVH * (S / WIN), KC, 0, stream>>>(qkv, out_kq);
  k_quant_v_token<<<KVH * S, VC, 0, stream>>>(qkv, out_vq);
  // 6) flash attention (one wave per head x 16-row q tile)
  {
    int waves = NH * (S / 16);                              // 4096
    k_flash_attn<<<waves / 8, 256, 0, stream>>>(q_bf, k_bf, vT_bf, at_bf);
  }
  // 7) output projection: [S x Hd] x [Hd x Hd] -> d_out f32
  {
    int tiles = (S / 32) * (Hd / 64);                       // 4096
    k_gemm_bf16<<<tiles / 8, 256, 0, stream>>>(at_bf, Wo_t, out_attn, S, Hd, Hd);
  }
}
